// Transformer_31808527794842
// MI455X (gfx1250) — compile-verified, matrix operand order checked by reading, not run on hardware
//
#include <hip/hip_runtime.h>
#include <hip/hip_bf16.h>

// ---------------------------------------------------------------------------
// Types for CDNA5 WMMA
// ---------------------------------------------------------------------------
typedef __attribute__((ext_vector_type(16))) __bf16 bf16x16;
typedef __attribute__((ext_vector_type(8)))  __bf16 bf16x8;
typedef __attribute__((ext_vector_type(8)))  float  f32x8;

#define BM 128
#define BN 128
#define BK 32
#define KPITCH 36   // f32 staging: BK + 4 pad -> 144B pitch (16B aligned)
#define BPITCH 40   // bf16 tiles:  BK + 8 pad -> 80B pitch (16B aligned)

// ---------------------------------------------------------------------------
// GEMM: C[M,N] = A[M,K](row major) * B[N,K](row major, i.e. B^T) (+epilogue)
// Pipeline per K-tile:
//   1) CDNA5 async DMA  global f32 -> LDS (ASYNCcnt, double buffered)
//   2) one conversion pass f32 -> compact bf16 LDS tiles (each element once)
//   3) fragments = plain 16B bf16 ds_loads -> v_wmma_f32_16x16x32_bf16
// Batched via blockIdx.z with split outer/inner strides.
// ---------------------------------------------------------------------------
struct GemmP {
  const float* A; const float* Bm; float* C;
  const float* bias; const float* res;
  int M, N, K;
  int lda, ldb, ldc;
  int zInner;                       // offset = (z/zInner)*out + (z%zInner)*in
  long long aOut, aIn, bOut, bIn, cOut, cIn;
  int flags;                        // 1=bias, 2=residual add, 4=exact gelu
};

__device__ __forceinline__ void async_b128(unsigned lds_off, const float* gptr) {
  // LDS dest byte offset in VGPR, 64-bit global address in VGPR pair.
  asm volatile("global_load_async_to_lds_b128 %0, %1, off"
               :: "v"(lds_off), "v"(gptr) : "memory");
}

__device__ __forceinline__ bf16x16 frag_from_lds(const __bf16* rowp, int kb) {
  // lane-local 16 K-values with the CDNA5 16-bit operand K half-swizzle:
  // elems 0..7 -> K = kb..kb+7, elems 8..15 -> K = 16+kb..16+kb+7
  bf16x8 lo = *(const bf16x8*)(rowp + kb);
  bf16x8 hi = *(const bf16x8*)(rowp + 16 + kb);
  return __builtin_shufflevector(lo, hi, 0,1,2,3,4,5,6,7,8,9,10,11,12,13,14,15);
}

__global__ __launch_bounds__(256) void gemm_bf16_wmma(GemmP p) {
  __shared__ float  Af[2][BM * KPITCH];   // async f32 landing pads, 2 x 18KB
  __shared__ float  Bf[2][BN * KPITCH];   // 2 x 18KB
  __shared__ __bf16 Ab[BM * BPITCH];      // compact bf16 tiles, 10KB each
  __shared__ __bf16 Bb[BN * BPITCH];

  const int z = blockIdx.z;
  const float* Ag = p.A  + (long long)(z / p.zInner) * p.aOut + (long long)(z % p.zInner) * p.aIn;
  const float* Bg = p.Bm + (long long)(z / p.zInner) * p.bOut + (long long)(z % p.zInner) * p.bIn;
  float*       Cg = p.C  + (long long)(z / p.zInner) * p.cOut + (long long)(z % p.zInner) * p.cIn;

  const int tileM = blockIdx.y * BM;
  const int tileN = blockIdx.x * BN;
  const int tid  = threadIdx.x;
  const int lane = tid & 31;
  const int wave = tid >> 5;          // 8 waves: 2 (M) x 4 (N)
  const int wm   = (wave >> 2) * 64;  // wave M offset in tile
  const int wn   = (wave & 3) * 32;   // wave N offset in tile
  const int lrow = lane & 15;
  const int kb   = (lane >> 4) * 8;   // K half-swizzle select

  f32x8 acc[4][2] = {};               // 4 M-frags x 2 N-frags of 16x16 f32

  const int ldr  = tid >> 3;          // staging: 32 rows/pass, 8 thr/row
  const int ldc4 = (tid & 7) * 4;     // 4 floats (16B) per thread per row

  // Issue one K-tile's async copies: 4 b128 per matrix per thread (8 total).
  // Row indices clamped instead of guarded: no faults, no divergence; clamped
  // rows only feed outputs discarded by the epilogue's col<N guard.
  auto issueA = [&](int buf, int k0) {
    #pragma unroll
    for (int r = 0; r < 4; ++r) {
      int rr = ldr + r * 32;
      int gr = tileM + rr; gr = gr < p.M ? gr : p.M - 1;
      const float* src = Ag + (long long)gr * p.lda + k0 + ldc4;
      async_b128((unsigned)(uintptr_t)&Af[buf][rr * KPITCH + ldc4], src);
    }
  };
  auto issueB = [&](int buf, int k0) {
    #pragma unroll
    for (int r = 0; r < 4; ++r) {
      int rr = ldr + r * 32;
      int gr = tileN + rr; gr = gr < p.N ? gr : p.N - 1;
      const float* src = Bg + (long long)gr * p.ldb + k0 + ldc4;
      async_b128((unsigned)(uintptr_t)&Bf[buf][rr * KPITCH + ldc4], src);
    }
  };

  const int ntiles = p.K / BK;        // K is always a multiple of 32 here
  issueA(0, 0);
  issueB(0, 0);

  for (int kt = 0; kt < ntiles; ++kt) {
    const int cur = kt & 1;
    if (kt + 1 < ntiles) {
      issueA(cur ^ 1, (kt + 1) * BK);
      issueB(cur ^ 1, (kt + 1) * BK);
      // The 8 outstanding copies belong to the *next* tile; in-order
      // completion means tile kt is fully resident in LDS.
      asm volatile("s_wait_asynccnt 0x8" ::: "memory");
    } else {
      asm volatile("s_wait_asynccnt 0x0" ::: "memory");
    }
    __syncthreads();   // tile kt f32 resident; iter kt-1 fragment reads done

    // ---- one-shot f32 -> bf16 conversion into compact tiles ----
    #pragma unroll
    for (int r = 0; r < 4; ++r) {
      int rr = ldr + r * 32;
      float4 fa = *(const float4*)&Af[cur][rr * KPITCH + ldc4];
      __bf16* da = &Ab[rr * BPITCH + ldc4];
      da[0] = (__bf16)fa.x; da[1] = (__bf16)fa.y; da[2] = (__bf16)fa.z; da[3] = (__bf16)fa.w;
      float4 fb = *(const float4*)&Bf[cur][rr * KPITCH + ldc4];
      __bf16* db = &Bb[rr * BPITCH + ldc4];
      db[0] = (__bf16)fb.x; db[1] = (__bf16)fb.y; db[2] = (__bf16)fb.z; db[3] = (__bf16)fb.w;
    }
    __syncthreads();   // bf16 tiles ready

    // ---- fragments + WMMA (pure bf16 ds_loads, no VALU in the chain) ----
    bf16x16 bfrag[2];
    #pragma unroll
    for (int j = 0; j < 2; ++j)
      bfrag[j] = frag_from_lds(&Bb[(wn + j * 16 + lrow) * BPITCH], kb);

    #pragma unroll
    for (int i = 0; i < 4; ++i) {
      bf16x16 af = frag_from_lds(&Ab[(wm + i * 16 + lrow) * BPITCH], kb);
      #pragma unroll
      for (int j = 0; j < 2; ++j)
        acc[i][j] = __builtin_amdgcn_wmma_f32_16x16x32_bf16(
            false, af, false, bfrag[j], (short)0, acc[i][j], false, false);
    }
  }

  // ---- epilogue: C/D layout — VGPR v -> row (v | v+8 by lane half), col = lane&15
  #pragma unroll
  for (int i = 0; i < 4; ++i) {
    #pragma unroll
    for (int j = 0; j < 2; ++j) {
      union { f32x8 v; float f[8]; } u; u.v = acc[i][j];
      #pragma unroll
      for (int v = 0; v < 8; ++v) {
        int row = tileM + wm + i * 16 + (lane >> 4) * 8 + v;
        int col = tileN + wn + j * 16 + lrow;
        if (row < p.M && col < p.N) {
          float xv = u.f[v];
          if (p.flags & 1) xv += p.bias[col];
          if (p.flags & 4) xv = 0.5f * xv * (1.0f + erff(xv * 0.70710678118654752f));
          if (p.flags & 2) xv += p.res[(long long)row * p.ldc + col];
          Cg[(long long)row * p.ldc + col] = xv;
        }
      }
    }
  }
}

// ---------------------------------------------------------------------------
// RMSNorm: y = x * rsqrt(mean(x^2)+eps) * g    (one row per block)
// ---------------------------------------------------------------------------
__global__ __launch_bounds__(256) void rmsnorm_kernel(const float* __restrict__ x,
                                                      const float* __restrict__ g,
                                                      float* __restrict__ y, int D) {
  __shared__ float red[256];
  long long row = blockIdx.x;
  const float* xr = x + row * D;
  float* yr = y + row * D;
  int tid = threadIdx.x;
  float acc = 0.f;
  for (int i = tid; i < D; i += 256) { float v = xr[i]; acc += v * v; }
  red[tid] = acc; __syncthreads();
  #pragma unroll
  for (int s = 128; s > 0; s >>= 1) {
    if (tid < s) red[tid] += red[tid + s];
    __syncthreads();
  }
  float r = rsqrtf(red[0] / (float)D + 1e-12f);
  for (int i = tid; i < D; i += 256) yr[i] = xr[i] * r * g[i];
}

// ---------------------------------------------------------------------------
// QKV split: qkv[B,T,3*H*DH] -> q[B,H,T,DH]*0.125, k[B,H,T,DH], vT[B,H,DH,T]
// ---------------------------------------------------------------------------
__global__ void qkv_split(const float* __restrict__ qkv, float* __restrict__ q,
                          float* __restrict__ k, float* __restrict__ vt,
                          int B, int T, int H, int DH) {
  long long idx = (long long)blockIdx.x * 256 + threadIdx.x;
  long long total = (long long)B * T * H * DH;
  if (idx >= total) return;
  int d = idx % DH; long long r = idx / DH;
  int h = r % H; r /= H;
  int t = r % T; int b = (int)(r / T);
  const float* src = qkv + ((long long)b * T + t) * (3 * H * DH);
  long long qo = (((long long)b * H + h) * T + t) * DH + d;
  q[qo]  = src[h * DH + d] * 0.125f;                 // fold 1/sqrt(64)
  k[qo]  = src[H * DH + h * DH + d];
  vt[(((long long)b * H + h) * DH + d) * T + t] = src[2 * H * DH + h * DH + d];
}

// ---------------------------------------------------------------------------
// Softmax + ALiBi, in place. One (b,h,row) per block. T==1024 -> 4 elems/thr.
// alibi = slope_h*|q-k| for q<T-nnb && k<T-nnb else 0;  slope_h = -2^(-(h+1)/2)
// ---------------------------------------------------------------------------
__global__ __launch_bounds__(256) void softmax_alibi(float* __restrict__ scores,
                                                     int T, int H, int nnb) {
  __shared__ float red[256];
  long long ridx = blockIdx.x;                // over B*H*T rows
  int r = (int)(ridx % T);
  int h = (int)((ridx / T) % H);
  float* row = scores + ridx * (long long)T;
  float slope = -exp2f(-0.5f * (float)(h + 1));
  int tid = threadIdx.x;
  float vals[4];
  float mx = -1e30f;
  #pragma unroll
  for (int it = 0; it < 4; ++it) {
    int c = tid + it * 256;
    float a = ((r < T - nnb) && (c < T - nnb)) ? slope * fabsf((float)(r - c)) : 0.0f;
    float v = row[c] + a;
    vals[it] = v;
    mx = fmaxf(mx, v);
  }
  red[tid] = mx; __syncthreads();
  #pragma unroll
  for (int s = 128; s > 0; s >>= 1) {
    if (tid < s) red[tid] = fmaxf(red[tid], red[tid + s]);
    __syncthreads();
  }
  mx = red[0]; __syncthreads();
  float sum = 0.f;
  #pragma unroll
  for (int it = 0; it < 4; ++it) { vals[it] = __expf(vals[it] - mx); sum += vals[it]; }
  red[tid] = sum; __syncthreads();
  #pragma unroll
  for (int s = 128; s > 0; s >>= 1) {
    if (tid < s) red[tid] += red[tid + s];
    __syncthreads();
  }
  float inv = 1.0f / red[0];
  #pragma unroll
  for (int it = 0; it < 4; ++it) row[tid + it * 256] = vals[it] * inv;
}

// ---------------------------------------------------------------------------
// concat along feature dim: c[row, 0:D]=a, c[row, D:2D]=b
// ---------------------------------------------------------------------------
__global__ void concat2(const float* __restrict__ a, const float* __restrict__ b,
                        float* __restrict__ c, long long rows, int D) {
  long long i = (long long)blockIdx.x * 256 + threadIdx.x;
  long long total = rows * 2 * D;
  if (i >= total) return;
  int col = (int)(i % (2 * D));
  long long row = i / (2 * D);
  c[i] = (col < D) ? a[row * D + col] : b[row * D + col - D];
}

// ---------------------------------------------------------------------------
// Host orchestration
// ---------------------------------------------------------------------------
static inline GemmP mk(const float* A, const float* Bm, float* C,
                       int M, int N, int K, int lda, int ldb, int ldc,
                       int flags, const float* bias = nullptr, const float* res = nullptr) {
  GemmP p{}; p.A = A; p.Bm = Bm; p.C = C; p.bias = bias; p.res = res;
  p.M = M; p.N = N; p.K = K; p.lda = lda; p.ldb = ldb; p.ldc = ldc;
  p.zInner = 1; p.aOut = p.aIn = p.bOut = p.bIn = p.cOut = p.cIn = 0;
  p.flags = flags; return p;
}

static inline void launch_gemm(const GemmP& p, int batch, hipStream_t s) {
  dim3 grid((p.N + BN - 1) / BN, (p.M + BM - 1) / BM, batch);
  gemm_bf16_wmma<<<grid, dim3(256), 0, s>>>(p);
}

extern "C" void kernel_launch(void* const* d_in, const int* in_sizes, int n_in,
                              void* d_out, int out_size, void* d_ws, size_t ws_size,
                              hipStream_t stream) {
  (void)in_sizes; (void)n_in; (void)out_size; (void)ws_size;
  const int B = 2, T = 1024, D = 1024, H = 16, DH = 64, FF = 4096, L = 8, NNB = 1;
  const int half = L / 2;
  const long long BT = (long long)B * T;

  const float* in_x   = (const float*)d_in[0];
  const float* ln1_g  = (const float*)d_in[1];
  const float* wqkv   = (const float*)d_in[2];
  const float* wo     = (const float*)d_in[3];
  const float* ln2_g  = (const float*)d_in[4];
  const float* w1     = (const float*)d_in[5];
  const float* b1     = (const float*)d_in[6];
  const float* w2     = (const float*)d_in[7];
  const float* b2     = (const float*)d_in[8];
  const float* skip_w = (const float*)d_in[9];
  const float* skip_b = (const float*)d_in[10];
  const float* out_g  = (const float*)d_in[11];

  char* base = (char*)d_ws;
  size_t off = 0;
  auto alloc = [&](size_t bytes) -> float* {
    float* p = (float*)(base + off);
    off += (bytes + 255) & ~(size_t)255;
    return p;
  };
  float* x      = alloc(BT * D * 4);
  float* xn     = alloc(BT * D * 4);
  float* qkv    = alloc(BT * 3 * H * DH * 4);
  float* q      = alloc(BT * D * 4);
  float* k      = alloc(BT * D * 4);
  float* vt     = alloc(BT * D * 4);
  float* scores = alloc((long long)B * H * T * T * 4);   // 128 MB
  float* yb     = alloc(BT * D * 4);
  float* ff     = alloc(BT * FF * 4);
  float* xc     = alloc(BT * 2 * D * 4);
  float* skipbuf[5];
  for (int i = 0; i < 5; ++i) skipbuf[i] = alloc(BT * D * 4);

  hipMemcpyAsync(x, in_x, BT * D * 4, hipMemcpyDeviceToDevice, stream);

  float* stackBuf[5]; int top = 0;
  for (int i = 0; i < L; ++i) {
    // ---- skip projection (layers 5..7): x = [x, pop()] @ skip_w[j]^T + skip_b[j]
    if (i > L - half) {
      float* s = stackBuf[--top];
      long long n = BT * 2 * D;
      concat2<<<dim3((unsigned)((n + 255) / 256)), 256, 0, stream>>>(x, s, xc, BT, D);
      int j = i - half;
      GemmP p = mk(xc, skip_w + (long long)j * D * 2 * D, x,
                   (int)BT, D, 2 * D, 2 * D, 2 * D, D, /*bias*/1,
                   skip_b + (long long)j * D);
      launch_gemm(p, 1, stream);
    }

    // ---- attention ----
    rmsnorm_kernel<<<dim3((unsigned)BT), 256, 0, stream>>>(x, ln1_g + (long long)i * D, xn, D);

    GemmP pq = mk(xn, wqkv + (long long)i * 3 * H * DH * D, qkv,
                  (int)BT, 3 * H * DH, D, D, D, 3 * H * DH, 0);
    launch_gemm(pq, 1, stream);

    long long nsplit = (long long)B * H * T * DH;
    qkv_split<<<dim3((unsigned)((nsplit + 255) / 256)), 256, 0, stream>>>(qkv, q, k, vt, B, T, H, DH);

    // scores[b,h] = q[b,h] @ k[b,h]^T   (scale folded into q)
    GemmP ps = mk(q, k, scores, T, T, DH, DH, DH, T, 0);
    ps.aOut = (long long)T * DH; ps.bOut = (long long)T * DH; ps.cOut = (long long)T * T;
    launch_gemm(ps, B * H, stream);

    softmax_alibi<<<dim3((unsigned)(B * H * T)), 256, 0, stream>>>(scores, T, H, NNB);

    // y[b,:,h*DH:(h+1)*DH] = att[b,h] @ v[b,h]   (vt is [B,H,DH,T])
    GemmP pv = mk(scores, vt, yb, T, DH, T, T, T, D, 0);
    pv.zInner = H;
    pv.aOut = (long long)H * T * T; pv.aIn = (long long)T * T;
    pv.bOut = (long long)H * DH * T; pv.bIn = (long long)DH * T;
    pv.cOut = (long long)T * D;      pv.cIn = DH;
    launch_gemm(pv, B * H, stream);

    // x = x + y @ wo^T
    GemmP po = mk(yb, wo + (long long)i * D * (H * DH), x,
                  (int)BT, D, H * DH, H * DH, H * DH, D, /*res*/2, nullptr, x);
    launch_gemm(po, 1, stream);

    // ---- mlp ----
    rmsnorm_kernel<<<dim3((unsigned)BT), 256, 0, stream>>>(x, ln2_g + (long long)i * D, xn, D);

    GemmP p1 = mk(xn, w1 + (long long)i * FF * D, ff,
                  (int)BT, FF, D, D, D, FF, /*bias|gelu*/1 | 4, b1 + (long long)i * FF);
    launch_gemm(p1, 1, stream);

    GemmP p2 = mk(ff, w2 + (long long)i * D * FF, x,
                  (int)BT, D, FF, FF, FF, D, /*bias|res*/1 | 2, b2 + (long long)i * D, x);
    launch_gemm(p2, 1, stream);

    // ---- push skip connection (layers 0..4) ----
    if (i <= half) {
      hipMemcpyAsync(skipbuf[i], x, BT * D * 4, hipMemcpyDeviceToDevice, stream);
      stackBuf[top++] = skipbuf[i];
    }
  }

  rmsnorm_kernel<<<dim3((unsigned)BT), 256, 0, stream>>>(x, out_g, (float*)d_out, D);
}
